// GATEncoder_6828998001487
// MI455X (gfx1250) — matrix operand order repaired and checked by reading, not compiled, not run on hardware
//
#include <hip/hip_runtime.h>
#include <hip/hip_bf16.h>
#include <math.h>

// ---------------- problem constants (from reference) ----------------
#define N_NODES 50000
#define N_EDGES 800000
#define N_GRAPH 512
#define DIN     64
#define HID     64
#define DOUT    128
#define NLAYER  3
#define DE      16
#define NEG_SLOPE 0.2f
#define BN_EPS    1e-5f

// ordered-uint encoding of -inf (for atomic float max via atomicMax(u32))
#define AMAX_INIT 0x007FFFFFu

typedef __attribute__((ext_vector_type(16))) _Float16 v16h;
typedef __attribute__((ext_vector_type(8)))  float    v8f;

// ---------------- monotonic float<->uint mapping for atomic max -----
__device__ __forceinline__ unsigned f2o(float f) {
    unsigned u = __float_as_uint(f);
    return (u & 0x80000000u) ? ~u : (u | 0x80000000u);
}
__device__ __forceinline__ float o2f(unsigned u) {
    return (u & 0x80000000u) ? __uint_as_float(u & 0x7fffffffu)
                             : __uint_as_float(~u);
}

// ---------------- WMMA helpers (CDNA5 16x16x32 f16 -> f32) ----------
__device__ __forceinline__ v8f wmma_f16(v16h a, v16h b, v8f c) {
    // 8 args: (neg_a, A, neg_b, B, c_mod, C, reuse_a, reuse_b)
    return __builtin_amdgcn_wmma_f32_16x16x32_f16(
        false, a, false, b, (short)0, c, false, false);
}

// A fragment: 16x32 (MxK) from row-major f32 [.,ld]; ISA 7.12.2 layout:
// lane&15 = row M; lanes 0-15 hold K {0..7,16..23}, lanes 16-31 K {8..15,24..31}
__device__ __forceinline__ v16h load_a_frag(const float* __restrict__ A,
                                            int row0, int ld, int kbase, int lane) {
    const float* p = A + (size_t)(row0 + (lane & 15)) * ld + kbase;
    int kh = ((lane >> 4) & 1) * 8;
    v16h a;
#pragma unroll
    for (int i = 0; i < 16; ++i) {
        int v = i >> 1;
        int k = (v < 4) ? (2 * v + (i & 1) + kh)
                        : (16 + 2 * (v - 4) + (i & 1) + kh);
        a[i] = (_Float16)p[k];
    }
    return a;
}

// B fragment: 32x16 (KxN) where B[k][n] = W[n][k] (out = in @ W^T),
// W row-major [nout][K].  lane&15 = col N; lanes 0-15 K {0..7,16..23}, lanes 16-31 K {8..15,24..31}.
__device__ __forceinline__ v16h load_b_frag_wt(const float* __restrict__ W,
                                               int ncol0, int ldk, int kbase, int lane) {
    const float* p = W + (size_t)(ncol0 + (lane & 15)) * ldk
                       + kbase + ((lane >> 4) & 1) * 8;
    v16h b;
#pragma unroll
    for (int i = 0; i < 16; ++i) b[i] = (_Float16)p[(i < 8) ? i : (i + 8)];
    return b;
}

// ---------------- dense transform: out[M,nout] = in[M,64] @ W^T + bias
// block = 128 threads (4 waves); wave w handles column tile blockIdx.y*4+w.
// M = gridDim.x*16 exactly (50000 = 3125*16, no remainder -> EXEC all-1s for WMMA)
__global__ void k_dense64(const float* __restrict__ in, const float* __restrict__ W,
                          const float* __restrict__ bias, float* __restrict__ out,
                          int nout) {
    int lane = threadIdx.x & 31;
    int wave = threadIdx.x >> 5;
    int row0  = blockIdx.x * 16;
    int ncol0 = (blockIdx.y * 4 + wave) * 16;

    v16h a0 = load_a_frag(in, row0, 64, 0,  lane);
    v16h a1 = load_a_frag(in, row0, 64, 32, lane);
    v16h b0 = load_b_frag_wt(W, ncol0, 64, 0,  lane);
    v16h b1 = load_b_frag_wt(W, ncol0, 64, 32, lane);

    float bv = bias[ncol0 + (lane & 15)];
    v8f c;
#pragma unroll
    for (int r = 0; r < 8; ++r) c[r] = bv;

    c = wmma_f16(a0, b0, c);
    c = wmma_f16(a1, b1, c);

    int n = ncol0 + (lane & 15);
    int mh = (lane >> 4) * 8;
#pragma unroll
    for (int r = 0; r < 8; ++r)
        out[(size_t)(row0 + mh + r) * nout + n] = c[r];
}

// ---------------- final linear (64->128) fused with graph-sum pooling
__global__ void k_lin_pool(const float* __restrict__ h, const float* __restrict__ W,
                           const float* __restrict__ bias, const int* __restrict__ batch,
                           float* __restrict__ pooled) {
    int lane = threadIdx.x & 31;
    int wave = threadIdx.x >> 5;
    int row0  = blockIdx.x * 16;
    int ncol0 = (blockIdx.y * 4 + wave) * 16;

    v16h a0 = load_a_frag(h, row0, 64, 0,  lane);
    v16h a1 = load_a_frag(h, row0, 64, 32, lane);
    v16h b0 = load_b_frag_wt(W, ncol0, 64, 0,  lane);
    v16h b1 = load_b_frag_wt(W, ncol0, 64, 32, lane);

    float bv = bias[ncol0 + (lane & 15)];
    v8f c;
#pragma unroll
    for (int r = 0; r < 8; ++r) c[r] = bv;
    c = wmma_f16(a0, b0, c);
    c = wmma_f16(a1, b1, c);

    int n  = ncol0 + (lane & 15);
    int mh = (lane >> 4) * 8;
#pragma unroll
    for (int r = 0; r < 8; ++r) {
        int row = row0 + mh + r;
        atomicAdd(&pooled[(size_t)batch[row] * DOUT + n], c[r]);
    }
}

// ---------------- per-layer scratch init --------------------------------
__global__ void k_init_layer(float* __restrict__ agg, unsigned* __restrict__ amax,
                             float* __restrict__ denom) {
    int i = blockIdx.x * blockDim.x + threadIdx.x;
    if (i < N_NODES * HID) agg[i] = 0.f;
    if (i < N_NODES) { amax[i] = AMAX_INIT; denom[i] = 0.f; }
}

// ---------------- edge pass A: attention logits + segment max -----------
// one wave per edge; lane handles channels {2*lane, 2*lane+1}
__global__ void k_edge_logits(const float* __restrict__ xl, const float* __restrict__ xr,
                              const float* __restrict__ eattr,
                              const int* __restrict__ src, const int* __restrict__ dst,
                              const float* __restrict__ We, const float* __restrict__ att,
                              float* __restrict__ alpha, unsigned* __restrict__ amax) {
    int lane = threadIdx.x & 31;
    int e = (blockIdx.x * blockDim.x + threadIdx.x) >> 5;
    if (e >= N_EDGES) return;
    int s = src[e], d = dst[e];
    int ch = lane * 2;

    float2 vl = *(const float2*)(xl + (size_t)s * HID + ch);
    float2 vr = *(const float2*)(xr + (size_t)d * HID + ch);

    // ea = We @ edge_attr[e]  (recomputed in-register; avoids 205 MB scratch)
    const float* eat = eattr + (size_t)e * DE;
    float ea0 = 0.f, ea1 = 0.f;
#pragma unroll
    for (int j = 0; j < DE; ++j) {
        float aj = eat[j];                // broadcast across wave
        ea0 = fmaf(We[(ch)     * DE + j], aj, ea0);
        ea1 = fmaf(We[(ch + 1) * DE + j], aj, ea1);
    }
    float m0 = vl.x + vr.x + ea0;
    float m1 = vl.y + vr.y + ea1;
    m0 = m0 > 0.f ? m0 : m0 * NEG_SLOPE;
    m1 = m1 > 0.f ? m1 : m1 * NEG_SLOPE;
    float p = fmaf(m0, att[ch], m1 * att[ch + 1]);
#pragma unroll
    for (int off = 16; off > 0; off >>= 1) p += __shfl_down(p, off);
    if (lane == 0) {
        alpha[e] = p;
        atomicMax(amax + d, f2o(p));
    }
}

// ---------------- edge pass B: exp + denom -------------------------------
__global__ void k_edge_exp(float* __restrict__ alpha, const int* __restrict__ dst,
                           const unsigned* __restrict__ amax, float* __restrict__ denom) {
    int e = blockIdx.x * blockDim.x + threadIdx.x;
    if (e >= N_EDGES) return;
    int d = dst[e];
    float ex = expf(alpha[e] - o2f(amax[d]));
    alpha[e] = ex;                         // reuse buffer for exp values
    atomicAdd(denom + d, ex);
}

// ---------------- edge pass C: weighted scatter-add ----------------------
__global__ void k_edge_agg(const float* __restrict__ xl, const float* __restrict__ eexp,
                           const float* __restrict__ denom,
                           const int* __restrict__ src, const int* __restrict__ dst,
                           float* __restrict__ agg) {
    int lane = threadIdx.x & 31;
    int e = (blockIdx.x * blockDim.x + threadIdx.x) >> 5;
    if (e >= N_EDGES) return;
    int s = src[e], d = dst[e];
    float w = eexp[e] / (denom[d] + 1e-16f);
    int ch = lane * 2;
    float2 vl = *(const float2*)(xl + (size_t)s * HID + ch);
    atomicAdd(agg + (size_t)d * HID + ch,     w * vl.x);
    atomicAdd(agg + (size_t)d * HID + ch + 1, w * vl.y);
}

// ---------------- conv-bias + BN + exact GELU + residual -----------------
__global__ void k_post(const float* __restrict__ agg, const float* __restrict__ cbias,
                       const float* __restrict__ gamma, const float* __restrict__ beta,
                       const float* __restrict__ mean,  const float* __restrict__ var,
                       float* __restrict__ h) {
    int i = blockIdx.x * blockDim.x + threadIdx.x;
    if (i >= N_NODES * HID) return;
    int ch = i & (HID - 1);
    float v = agg[i] + cbias[ch];
    v = (v - mean[ch]) * rsqrtf(var[ch] + BN_EPS) * gamma[ch] + beta[ch];
    float g = 0.5f * v * (1.f + erff(v * 0.70710678118654752f));
    h[i] += g;
}

// ---------------- pooling helpers ----------------------------------------
__global__ void k_init_pool(float* __restrict__ pooled, float* __restrict__ counts) {
    int i = blockIdx.x * blockDim.x + threadIdx.x;
    if (i < N_GRAPH * DOUT) pooled[i] = 0.f;
    if (i < N_GRAPH) counts[i] = 0.f;
}
__global__ void k_counts(const int* __restrict__ batch, float* __restrict__ counts) {
    int i = blockIdx.x * blockDim.x + threadIdx.x;
    if (i < N_NODES) atomicAdd(counts + batch[i], 1.f);
}
__global__ void k_div(float* __restrict__ pooled, const float* __restrict__ counts) {
    int i = blockIdx.x * blockDim.x + threadIdx.x;
    if (i >= N_GRAPH * DOUT) return;
    pooled[i] /= fmaxf(counts[i / DOUT], 1.f);
}

// ---------------- host launcher ------------------------------------------
extern "C" void kernel_launch(void* const* d_in, const int* in_sizes, int n_in,
                              void* d_out, int out_size, void* d_ws, size_t ws_size,
                              hipStream_t stream) {
    (void)in_sizes; (void)n_in; (void)out_size; (void)ws_size;

    const float* x          = (const float*)d_in[0];
    const int*   edge_index = (const int*)  d_in[1];
    const float* edge_attr  = (const float*)d_in[2];
    const int*   batch      = (const int*)  d_in[3];
    const float* emb_W      = (const float*)d_in[4];
    const float* emb_b      = (const float*)d_in[5];
    const float* Wl         = (const float*)d_in[6];
    const float* bl         = (const float*)d_in[7];
    const float* Wr         = (const float*)d_in[8];
    const float* br         = (const float*)d_in[9];
    const float* We         = (const float*)d_in[10];
    const float* att        = (const float*)d_in[11];
    const float* conv_bias  = (const float*)d_in[12];
    const float* bn_gamma   = (const float*)d_in[13];
    const float* bn_beta    = (const float*)d_in[14];
    const float* bn_mean    = (const float*)d_in[15];
    const float* bn_var     = (const float*)d_in[16];
    const float* lin_W      = (const float*)d_in[17];
    const float* lin_b      = (const float*)d_in[18];
    float* pooled = (float*)d_out;

    const int* src = edge_index;             // edge_index[0,:]
    const int* dst = edge_index + N_EDGES;   // edge_index[1,:]

    // carve workspace (256B-aligned slices)
    char* ws = (char*)d_ws;
    size_t off = 0;
    auto carve = [&](size_t bytes) -> void* {
        void* p = ws + off;
        off = (off + bytes + 255) & ~(size_t)255;
        return p;
    };
    float*    h      = (float*)   carve((size_t)N_NODES * HID * 4);
    float*    xl     = (float*)   carve((size_t)N_NODES * HID * 4);
    float*    xr     = (float*)   carve((size_t)N_NODES * HID * 4);
    float*    agg    = (float*)   carve((size_t)N_NODES * HID * 4);
    float*    alpha  = (float*)   carve((size_t)N_EDGES * 4);
    unsigned* amax   = (unsigned*)carve((size_t)N_NODES * 4);
    float*    denom  = (float*)   carve((size_t)N_NODES * 4);
    float*    counts = (float*)   carve((size_t)N_GRAPH * 4);

    const int MTILES = N_NODES / 16;               // 3125, exact
    dim3 gemm_grid(MTILES, 1), gemm_blk(128);      // 4 waves = 4 col tiles of 64
    int nc_blocks = (N_NODES * HID + 255) / 256;   // node*channel grids
    int ew_blocks = N_EDGES / 8;                   // wave-per-edge, 8 edges/block
    int e_blocks  = (N_EDGES + 255) / 256;

    // h = x @ emb_W^T + emb_b
    k_dense64<<<gemm_grid, gemm_blk, 0, stream>>>(x, emb_W, emb_b, h, HID);

    for (int l = 0; l < NLAYER; ++l) {
        const float* Wl_l = Wl + (size_t)l * HID * HID;
        const float* Wr_l = Wr + (size_t)l * HID * HID;
        const float* We_l = We + (size_t)l * HID * DE;

        k_dense64<<<gemm_grid, gemm_blk, 0, stream>>>(h, Wl_l, bl + l * HID, xl, HID);
        k_dense64<<<gemm_grid, gemm_blk, 0, stream>>>(h, Wr_l, br + l * HID, xr, HID);

        k_init_layer<<<nc_blocks, 256, 0, stream>>>(agg, amax, denom);
        k_edge_logits<<<ew_blocks, 256, 0, stream>>>(xl, xr, edge_attr, src, dst,
                                                     We_l, att + l * HID, alpha, amax);
        k_edge_exp<<<e_blocks, 256, 0, stream>>>(alpha, dst, amax, denom);
        k_edge_agg<<<ew_blocks, 256, 0, stream>>>(xl, alpha, denom, src, dst, agg);

        k_post<<<nc_blocks, 256, 0, stream>>>(agg, conv_bias + l * HID,
                                              bn_gamma + l * HID, bn_beta + l * HID,
                                              bn_mean + l * HID, bn_var + l * HID, h);
    }

    // pooled = segment_mean(h @ lin_W^T + lin_b, batch)
    int p_blocks = (N_GRAPH * DOUT + 255) / 256;
    k_init_pool<<<p_blocks, 256, 0, stream>>>(pooled, counts);
    k_counts<<<(N_NODES + 255) / 256, 256, 0, stream>>>(batch, counts);
    k_lin_pool<<<dim3(MTILES, DOUT / 64), gemm_blk, 0, stream>>>(h, lin_W, lin_b,
                                                                 batch, pooled);
    k_div<<<p_blocks, 256, 0, stream>>>(pooled, counts);
}
